// Heirachical_Loss_72808285602035
// MI455X (gfx1250) — matrix operand order repaired and checked by reading, not compiled, not run on hardware
//
#include <hip/hip_runtime.h>
#include <hip/hip_bf16.h>

// Hierarchical softmax loss, MI455X (gfx1250, wave32).
// Memory-bound: 85MB in / ~1MB out -> ~3.7us at 23.3 TB/s HBM peak.
// - Tile staging global->LDS uses the Tensor Data Mover (one tensor_load_to_lds
//   per wave: contiguous 5184B tile), tracked with TENSORcnt.
// - The (Bx81)@(81x121) ancestor-mask matmul runs on v_wmma_f32_16x16x32_f16
//   (0/1 mask exact in f16, f32 accumulation), only for the 40 internal-node
//   columns; leaf probs are the exp values already in LDS.
// - Tree topology (branch 3, offsets 0/81/108/117/120) is compile-time
//   constant: ancestors of leaf t are t/3, t/9, t/27 analytically.

typedef _Float16 v16h __attribute__((ext_vector_type(16)));
typedef _Float16 v8h  __attribute__((ext_vector_type(8)));
typedef float    v8f  __attribute__((ext_vector_type(8)));
typedef unsigned v4u  __attribute__((ext_vector_type(4)));
typedef int      v4i  __attribute__((ext_vector_type(4)));
typedef int      v8i  __attribute__((ext_vector_type(8)));

#define NW 4          // waves per block
#define NLEAF 81
#define KPAD 96       // 81 padded to 3 x 32 (WMMA K)
#define NNODE 48      // 40 internal nodes (81..120) padded to 3 x 16 (WMMA N)
#define TILE_DW (16 * NLEAF)   // 1296 dwords = 5184 B per wave tile

#ifndef __has_builtin
#define __has_builtin(x) 0
#endif
#if __has_builtin(__builtin_amdgcn_tensor_load_to_lds) && \
    __has_builtin(__builtin_amdgcn_s_wait_tensorcnt)
#define USE_TDM 1
#else
#define USE_TDM 0
#endif

__global__ void hsm_init_out(float* out, int B) {
    out[0] = 0.0f;       // loss accumulator
    out[B + 1] = 0.0f;   // total_dist accumulator
}

__global__ __launch_bounds__(128) void hsm_kernel(const float* __restrict__ x,
                                                  const int* __restrict__ tgt,
                                                  float* __restrict__ out,
                                                  int B) {
    // LDS: 9216 + 20736 + 12288 + 12288 = 54528 bytes
    __shared__ alignas(32) _Float16 maskBt[NNODE][KPAD];  // B^T: [n][k], 0/1
    __shared__ alignas(16) float stag[NW][16][NLEAF];     // staged logits (TDM dest)
    __shared__ alignas(32) _Float16 ex[NW][16][KPAD];     // exp(x - max), f16
    __shared__ float probsI[NW][16][NNODE];               // internal-node sums

    const int tid  = threadIdx.x;
    const int w    = tid >> 5;        // wave in block
    const int lane = tid & 31;
    const int g    = lane >> 4;       // half-wave group (WMMA layout)
    const int l15  = lane & 15;
    const int R    = (blockIdx.x * NW + w) * 16;  // first row of this wave's tile

#if USE_TDM
    // ---- TDM: one tensor_load_to_lds per wave for its contiguous 5184B tile ----
    {
        unsigned lds_off = (unsigned)(uintptr_t)&stag[w][0][0];  // addr[31:0] = LDS byte offset
        unsigned long long ga = (unsigned long long)(uintptr_t)(x + (size_t)R * NLEAF);
        lds_off         = __builtin_amdgcn_readfirstlane(lds_off);
        unsigned ga_lo  = __builtin_amdgcn_readfirstlane((unsigned)ga);
        unsigned ga_hi  = __builtin_amdgcn_readfirstlane((unsigned)(ga >> 32));

        // D# group0: count=1 | lds_addr | global_addr[56:0] | type=2 (bits 127:126)
        v4u g0 = { 1u, lds_off, ga_lo, (ga_hi & 0x01FFFFFFu) | 0x80000000u };
        // D# group1: data_size=2 (4B); tensor_dim0=tile_dim0=1296;
        // tensor_dim1=tile_dim1=1; tensor_dim0_stride=1296; rest 0.
        v8i g1 = { (int)0x00020000,
                   (int)((unsigned)TILE_DW << 16),            // abar=0 | tdim0.lo16
                   (int)((TILE_DW >> 16) | (1u << 16)),       // tdim0.hi16 | tdim1.lo16
                   (int)((unsigned)TILE_DW << 16),            // tdim1.hi16 | tile_dim0
                   1,                                         // tile_dim1=1, tile_dim2=0
                   (int)TILE_DW,                              // tdim0_stride.lo32
                   0, 0 };
        v4i z4 = { 0, 0, 0, 0 };
#if __clang_major__ >= 23
        v8i z8 = { 0, 0, 0, 0, 0, 0, 0, 0 };
        __builtin_amdgcn_tensor_load_to_lds(g0, g1, z4, z4, z8, 0);
#else
        __builtin_amdgcn_tensor_load_to_lds(g0, g1, z4, z4, 0);
#endif
    }
#endif

    // ---- Build constant 0/1 B-matrix (transposed, f16) from tree formulas ----
    // column n -> node 81+n: n<27 level1 (k/3==n), n<36 level2 (k/9==n-27),
    // n<39 level3 (k/27==n-36), n==39 root (all k<81), n>=40 zero pad.
    for (int i = tid; i < NNODE * KPAD; i += 128) {
        int n = i / KPAD, k = i - n * KPAD;
        float v = 0.0f;
        if (k < NLEAF) {
            if      (n < 27) v = (k / 3  == n)      ? 1.0f : 0.0f;
            else if (n < 36) v = (k / 9  == n - 27) ? 1.0f : 0.0f;
            else if (n < 39) v = (k / 27 == n - 36) ? 1.0f : 0.0f;
            else if (n == 39) v = 1.0f;
        }
        maskBt[n][k] = (_Float16)v;
    }

#if USE_TDM
    __builtin_amdgcn_s_wait_tensorcnt(0);  // this wave's tile is in LDS
#else
    // ---- Fallback: coalesced stage of 16 rows x 81 logits into LDS ----
    for (int i = lane; i < TILE_DW; i += 32) {
        int rr = i / NLEAF, cc = i - rr * NLEAF;
        stag[w][rr][cc] = x[(size_t)R * NLEAF + i];
        (void)rr; (void)cc;
    }
#endif
    __syncthreads();

    // ---- Per-row softmax numerators: e = exp(x - max) as f16 A-tile ----
    if (lane < 16) {
        float m = -3.4e38f;
        for (int k = 0; k < NLEAF; ++k) m = fmaxf(m, stag[w][lane][k]);
        for (int k = 0; k < NLEAF; ++k)
            ex[w][lane][k] = (_Float16)__expf(stag[w][lane][k] - m);
        for (int k = NLEAF; k < KPAD; ++k) ex[w][lane][k] = (_Float16)0.0f;
    }
    __syncthreads();

    // ---- WMMA: probsI(16x48) = E(16x96,f16) @ maskBt^T(96x48,f16), f32 acc ----
    // A layout (16-bit 16x32): lane L holds row L&15; contiguous
    // K=32kc+16g+{0..15} in elems 0..15 (matches B).
    v16h afrag[3];
#pragma unroll
    for (int kc = 0; kc < 3; ++kc) {
        v8h lo = *(const v8h*)&ex[w][l15][32 * kc + 16 * g];
        v8h hi = *(const v8h*)&ex[w][l15][32 * kc + 16 * g + 8];
        afrag[kc] = __builtin_shufflevector(lo, hi, 0, 1, 2, 3, 4, 5, 6, 7,
                                            8, 9, 10, 11, 12, 13, 14, 15);
    }
#pragma unroll
    for (int nt = 0; nt < 3; ++nt) {
        v8f c = {0.f, 0.f, 0.f, 0.f, 0.f, 0.f, 0.f, 0.f};
#pragma unroll
        for (int kc = 0; kc < 3; ++kc) {
            // B layout (16-bit 32x16): lane holds column l15; contiguous
            // K = 32kc+16g .. +15 in elems 0..15 (32B aligned LDS read).
            v16h b = *(const v16h*)&maskBt[nt * 16 + l15][32 * kc + 16 * g];
            c = __builtin_amdgcn_wmma_f32_16x16x32_f16(
                    false, afrag[kc], false, b, (short)0, c, false, false);
        }
        // C layout: VGPR j -> row j+8g, col = lane&15 of this n-tile.
#pragma unroll
        for (int j = 0; j < 8; ++j)
            probsI[w][j + 8 * g][nt * 16 + l15] = c[j];
    }
    __syncthreads();

    // ---- Per-row path loss + greedy descent + ancestor distance ----
    float loss_r = 0.0f, distv = 0.0f;
    if (lane < 16) {
        const float* P = probsI[w][lane];  // 0..26:lvl1  27..35:lvl2  36..38:lvl3  39:root(Z)
        int t = tgt[R + lane];
        float Z   = P[39];
        float et  = (float)ex[w][lane][t];
        float s1v = P[t / 3];
        float s2v = P[27 + t / 9];
        float s3v = P[36 + t / 27];
        // 2*(win-0.5) = (0.5*s3 + 0.25*s2 + 0.125*s1 + 0.125*e_t)/Z
        float num = 0.5f * s3v + 0.25f * s2v + 0.125f * s1v + 0.125f * et;
        loss_r = __logf(Z) - __logf(num);

        // greedy argmax descent; strict '>' matches jnp.argmax first-max rule
        int j3 = 0;
        { float b = P[36]; if (P[37] > b) { b = P[37]; j3 = 1; } if (P[38] > b) j3 = 2; }
        int j2 = 3 * j3;
        { float b = P[27 + 3 * j3];
          for (int c2 = 1; c2 < 3; ++c2) { float v = P[27 + 3 * j3 + c2]; if (v > b) { b = v; j2 = 3 * j3 + c2; } } }
        int j1 = 3 * j2;
        { float b = P[3 * j2];
          for (int c2 = 1; c2 < 3; ++c2) { float v = P[3 * j2 + c2]; if (v > b) { b = v; j1 = 3 * j2 + c2; } } }
        int leaf = 3 * j1;
        { float b = (float)ex[w][lane][3 * j1];
          for (int c2 = 1; c2 < 3; ++c2) { float v = (float)ex[w][lane][3 * j1 + c2]; if (v > b) { b = v; leaf = 3 * j1 + c2; } } }

        out[1 + R + lane] = (float)leaf;  // preds
        distv = (leaf == t) ? 0.0f
              : (leaf / 3  == t / 3 ) ? 1.0f
              : (leaf / 9  == t / 9 ) ? 2.0f
              : (leaf / 27 == t / 27) ? 3.0f : 4.0f;
    }

    // ---- wave32 reduction; lanes 16..31 contribute 0 ----
    for (int off = 16; off; off >>= 1) {
        loss_r += __shfl_xor(loss_r, off);
        distv  += __shfl_xor(distv, off);
    }
    if (lane == 0) {
        atomicAdd(&out[0], loss_r * (1.0f / (float)B));
        atomicAdd(&out[B + 1], distv);
    }
}

extern "C" void kernel_launch(void* const* d_in, const int* in_sizes, int n_in,
                              void* d_out, int out_size, void* d_ws, size_t ws_size,
                              hipStream_t stream) {
    (void)n_in; (void)out_size; (void)d_ws; (void)ws_size;
    const float* x  = (const float*)d_in[0];   // outputs (B, 81) f32
    const int* tgt  = (const int*)d_in[1];     // target (B,) i32
    float* out      = (float*)d_out;           // [loss, preds(B), total_dist]
    int B = in_sizes[1];

    hsm_init_out<<<1, 1, 0, stream>>>(out, B);
    // 4 waves/block, 16 rows/wave -> 64 rows/block. B = 262144 is divisible.
    int blocks = B / (NW * 16);
    hsm_kernel<<<blocks, 128, 0, stream>>>(x, tgt, out, B);
}
